// MultiHeadAttention_81372450390704
// MI455X (gfx1250) — compile-verified, matrix operand order checked by reading, not run on hardware
//
#include <hip/hip_runtime.h>

typedef __attribute__((ext_vector_type(16))) _Float16 v16h;
typedef __attribute__((ext_vector_type(8)))  float    v8f;
typedef __attribute__((ext_vector_type(4)))  int      v4i;

#define DM    512
#define SEQ   1024
#define BATCH 16
#define HEADS 8
#define HD    64
#define NB    (HEADS * BATCH)   /* 128 */
#define NEGC  (-4294967295.0f)

// K-index map for 16-bit A/B fragments (per ISA 7.12.2): per lane-half the
// 16 halves map to two contiguous 8-half chunks -> ds_load_b128 gathers.
__device__ __forceinline__ int kmap(int h, int half) {
    int v = h >> 1, p = h & 1;
    int k = (v < 4) ? (2 * v + p) : (16 + 2 * (v - 4) + p);
    return k + 8 * half;
}

// ---- async global->LDS staging (CDNA5), with safe fallback -----------------
#if __has_builtin(__builtin_amdgcn_global_load_async_to_lds_b128) && \
    __has_builtin(__builtin_amdgcn_s_wait_asynccnt)
#define HAVE_ASYNC_LDS 1
__device__ __forceinline__ void async_cp16(void* lds_dst, const void* gsrc) {
    __builtin_amdgcn_global_load_async_to_lds_b128(
        (__attribute__((address_space(1))) v4i*)gsrc,
        (__attribute__((address_space(3))) v4i*)lds_dst, 0, 0);
}
__device__ __forceinline__ void async_wait() {
    __builtin_amdgcn_s_wait_asynccnt(0);
}
#else
#define HAVE_ASYNC_LDS 0
__device__ __forceinline__ void async_cp16(void* lds_dst, const void* gsrc) {
    *(uint4*)lds_dst = *(const uint4*)gsrc;
}
__device__ __forceinline__ void async_wait() {}
#endif

__global__ void cvt_f32_f16(const float* __restrict__ src,
                            _Float16* __restrict__ dst, int n) {
    int i = blockIdx.x * blockDim.x + threadIdx.x;
    int stride = gridDim.x * blockDim.x;
    for (; i < n; i += stride) dst[i] = (_Float16)src[i];
}

// Transposing convert for 512x512 weights: dst[c][r] = src[r][c].
__global__ void cvt_t_f32_f16(const float* __restrict__ src,
                              _Float16* __restrict__ dst) {
    int i = blockIdx.x * blockDim.x + threadIdx.x;
    int stride = gridDim.x * blockDim.x;
    for (; i < DM * DM; i += stride) {
        int c = i >> 9, r = i & 511;
        dst[i] = (_Float16)src[r * DM + c];
    }
}

// ---------------------------------------------------------------------------
// QKV projection: Y = Xh(16384x512) @ W + b, W given TRANSPOSED ([cout][k]).
// Double-buffered async LDS pipeline; 4 waves; 64x64 block tile.
// ---------------------------------------------------------------------------
__global__ __launch_bounds__(128)
void gemm_qkv(const _Float16* __restrict__ xh,
              const _Float16* __restrict__ wqt,
              const _Float16* __restrict__ wkt,
              const _Float16* __restrict__ wvt,
              const float* __restrict__ bq,
              const float* __restrict__ bk,
              const float* __restrict__ bv,
              _Float16* __restrict__ Qh,
              _Float16* __restrict__ Kh,
              _Float16* __restrict__ VhT) {
    __shared__ __align__(16) _Float16 x_lds[2][64][32];   // [buf][row][k]
    __shared__ __align__(16) _Float16 w_lds[2][64][32];   // [buf][cout][k]

    const int mode = blockIdx.z;
    const _Float16* wtp = (mode == 0) ? wqt : ((mode == 1) ? wkt : wvt);
    const float*    bia = (mode == 0) ? bq  : ((mode == 1) ? bk  : bv);

    const int row0 = blockIdx.x * 64;
    const int cb   = blockIdx.y;
    const int t    = threadIdx.x;
    const int lane = t & 31, w = t >> 5;
    const int half = lane >> 4, l16 = lane & 15;

    const int rs = t >> 1, cs = (t & 1) * 16;     // staging coords (64x32 tile)

    auto stage = [&](int kb, int buf) {
        const _Float16* gx = xh + (size_t)(row0 + rs) * DM + kb * 32 + cs;
        _Float16* sx = &x_lds[buf][rs][cs];
        async_cp16(sx, gx); async_cp16(sx + 8, gx + 8);
        const _Float16* gw = wtp + (size_t)(cb * 64 + rs) * DM + kb * 32 + cs;
        _Float16* sw = &w_lds[buf][rs][cs];
        async_cp16(sw, gw); async_cp16(sw + 8, gw + 8);
    };

    v8f acc[4] = {};

    stage(0, 0);
    async_wait();
    __syncthreads();
    stage(1, 1);

    for (int kb = 0; kb < 16; ++kb) {
        const int buf = kb & 1;
        v16h a;
        #pragma unroll
        for (int h = 0; h < 16; ++h) a[h] = x_lds[buf][w * 16 + l16][kmap(h, half)];
        #pragma unroll
        for (int f = 0; f < 4; ++f) {
            v16h b;
            #pragma unroll
            for (int h = 0; h < 16; ++h) b[h] = w_lds[buf][f * 16 + l16][kmap(h, half)];
            acc[f] = __builtin_amdgcn_wmma_f32_16x16x32_f16(
                false, a, false, b, (short)0, acc[f], false, false);
        }
        if (kb < 15) {
            async_wait();          // tile kb+1 resident
            __syncthreads();       // all waves done with buffer kb&1
            if (kb < 14) stage(kb + 2, buf);
        }
    }

    // store head-major: col c -> head hh; row -> (b, s); uniform branch hoisted
    if (mode == 2) {
        #pragma unroll
        for (int f = 0; f < 4; ++f) {
            int c = cb * 64 + f * 16 + l16;
            int hh = c >> 6, within = c & 63;
            float bsv = bia[c];
            #pragma unroll
            for (int r = 0; r < 8; ++r) {
                int row = row0 + w * 16 + r + 8 * half;
                int b = row >> 10, s = row & 1023;
                int n = hh * 16 + b;
                VhT[((size_t)n * HD + within) * SEQ + s] = (_Float16)(acc[f][r] + bsv);
            }
        }
    } else {
        _Float16* dst = (mode == 0) ? Qh : Kh;
        #pragma unroll
        for (int f = 0; f < 4; ++f) {
            int c = cb * 64 + f * 16 + l16;
            int hh = c >> 6, within = c & 63;
            float bsv = bia[c];
            #pragma unroll
            for (int r = 0; r < 8; ++r) {
                int row = row0 + w * 16 + r + 8 * half;
                int b = row >> 10, s = row & 1023;
                int n = hh * 16 + b;
                dst[((size_t)n * SEQ + s) * HD + within] = (_Float16)(acc[f][r] + bsv);
            }
        }
    }
}

// ---------------------------------------------------------------------------
// Fused flash-style attention: per block = one n (h*B) x 64 query rows.
// Double-buffered async K/V pipeline; raw scores written exactly once.
// ---------------------------------------------------------------------------
__global__ __launch_bounds__(128)
void attn(const _Float16* __restrict__ Qg, const _Float16* __restrict__ Kg,
          const _Float16* __restrict__ VgT, const int* __restrict__ pmask,
          float* __restrict__ raw, _Float16* __restrict__ heads) {
    __shared__ __align__(16) _Float16 q_lds[64][64];       // [qrow][feat]
    __shared__ __align__(16) _Float16 k_lds[2][32][64];    // [buf][key][feat]
    __shared__ __align__(16) _Float16 v_lds[2][64][32];    // [buf][vcol][key]
    __shared__ __align__(16) _Float16 p_st[4][16][32];     // per-wave P staging

    const int n  = blockIdx.y;
    const int q0 = blockIdx.x * 64;
    const int t = threadIdx.x, lane = t & 31, w = t >> 5;
    const int half = lane >> 4, l16 = lane & 15;
    // faithful index mismatch: mask indexed batch-major (n -> b = n/H)
    const int* pmrow = pmask + (size_t)(n >> 3) * SEQ;

    const int rk = t >> 2, ck = (t & 3) * 16;   // K staging coords (32x64)
    const int rv = t >> 1, cv = (t & 1) * 16;   // V staging coords (64x32)

    auto stage = [&](int kb, int buf) {
        const _Float16* gk = Kg + ((size_t)n * SEQ + kb * 32 + rk) * HD + ck;
        _Float16* sk = &k_lds[buf][rk][ck];
        async_cp16(sk, gk); async_cp16(sk + 8, gk + 8);
        const _Float16* gv = VgT + ((size_t)n * HD + rv) * SEQ + kb * 32 + cv;
        _Float16* sv = &v_lds[buf][rv][cv];
        async_cp16(sv, gv); async_cp16(sv + 8, gv + 8);
    };

    {   // stage 64x64 Q tile alongside K/V tile 0
        int r = t >> 1, c = (t & 1) * 32;
        const _Float16* g = Qg + ((size_t)n * SEQ + q0 + r) * HD + c;
        _Float16* s = &q_lds[r][c];
        async_cp16(s,      g);
        async_cp16(s + 8,  g + 8);
        async_cp16(s + 16, g + 16);
        async_cp16(s + 24, g + 24);
    }
    stage(0, 0);
    async_wait();
    __syncthreads();

    v16h aq0, aq1;
    #pragma unroll
    for (int h = 0; h < 16; ++h) {
        aq0[h] = q_lds[w * 16 + l16][kmap(h, half)];
        aq1[h] = q_lds[w * 16 + l16][32 + kmap(h, half)];
    }

    stage(1, 1);   // prefetch tile 1 while computing tile 0

    v8f o[4] = {};
    float mrow[8], lrow[8];
    #pragma unroll
    for (int r = 0; r < 8; ++r) { mrow[r] = -3.0e38f; lrow[r] = 0.0f; }

    for (int kb = 0; kb < 32; ++kb) {
        const int buf = kb & 1;

        // S tile (16 q-rows x 32 keys): contraction over 64 features
        v8f s0 = {}, s1 = {};
        {
            v16h b;
            #pragma unroll
            for (int h = 0; h < 16; ++h) b[h] = k_lds[buf][l16][kmap(h, half)];
            s0 = __builtin_amdgcn_wmma_f32_16x16x32_f16(false, aq0, false, b, (short)0, s0, false, false);
            #pragma unroll
            for (int h = 0; h < 16; ++h) b[h] = k_lds[buf][l16][32 + kmap(h, half)];
            s0 = __builtin_amdgcn_wmma_f32_16x16x32_f16(false, aq1, false, b, (short)0, s0, false, false);
            #pragma unroll
            for (int h = 0; h < 16; ++h) b[h] = k_lds[buf][16 + l16][kmap(h, half)];
            s1 = __builtin_amdgcn_wmma_f32_16x16x32_f16(false, aq0, false, b, (short)0, s1, false, false);
            #pragma unroll
            for (int h = 0; h < 16; ++h) b[h] = k_lds[buf][16 + l16][32 + kmap(h, half)];
            s1 = __builtin_amdgcn_wmma_f32_16x16x32_f16(false, aq1, false, b, (short)0, s1, false, false);
        }

        // scale, emit raw score, apply mask
        int pm0 = pmrow[kb * 32 + l16];
        int pm1 = pmrow[kb * 32 + 16 + l16];
        float ms0[8], ms1[8];
        float* rawbase = raw + ((size_t)n * SEQ + q0 + w * 16) * SEQ + kb * 32;
        #pragma unroll
        for (int r = 0; r < 8; ++r) {
            float v0 = s0[r] * 0.125f, v1 = s1[r] * 0.125f;
            int rr = r + 8 * half;
            rawbase[(size_t)rr * SEQ + l16]      = v0;
            rawbase[(size_t)rr * SEQ + 16 + l16] = v1;
            ms0[r] = pm0 ? v0 : NEGC;
            ms1[r] = pm1 ? v1 : NEGC;
        }

        // online softmax; row M = r + 8*half lives across one 16-lane half
        #pragma unroll
        for (int r = 0; r < 8; ++r) {
            float tmax = fmaxf(ms0[r], ms1[r]);
            #pragma unroll
            for (int m = 1; m < 16; m <<= 1) tmax = fmaxf(tmax, __shfl_xor(tmax, m, 32));
            float mnew  = fmaxf(mrow[r], tmax);
            float scale = __expf(mrow[r] - mnew);
            float p0 = __expf(ms0[r] - mnew);
            float p1 = __expf(ms1[r] - mnew);
            float psum = p0 + p1;
            #pragma unroll
            for (int m = 1; m < 16; m <<= 1) psum += __shfl_xor(psum, m, 32);
            lrow[r] = lrow[r] * scale + psum;
            mrow[r] = mnew;
            #pragma unroll
            for (int f = 0; f < 4; ++f) o[f][r] *= scale;
            p_st[w][r + 8 * half][l16]      = (_Float16)p0;
            p_st[w][r + 8 * half][16 + l16] = (_Float16)p1;
        }

        // P(16x32) @ V(32x64), V staged transposed so B gathers are contiguous
        v16h ap;
        #pragma unroll
        for (int h = 0; h < 16; ++h) ap[h] = p_st[w][l16][kmap(h, half)];
        #pragma unroll
        for (int f = 0; f < 4; ++f) {
            v16h bvf;
            #pragma unroll
            for (int h = 0; h < 16; ++h) bvf[h] = v_lds[buf][f * 16 + l16][kmap(h, half)];
            o[f] = __builtin_amdgcn_wmma_f32_16x16x32_f16(
                false, ap, false, bvf, (short)0, o[f], false, false);
        }

        if (kb < 31) {
            async_wait();          // tile kb+1 resident
            __syncthreads();       // all waves done reading buffer kb&1
            if (kb < 30) stage(kb + 2, buf);
        }
    }

    // epilogue: normalize, write heads in (B, S, H*V) f16 layout
    const int hh = n >> 4, b = n & 15;
    #pragma unroll
    for (int f = 0; f < 4; ++f) {
        #pragma unroll
        for (int r = 0; r < 8; ++r) {
            int s = q0 + w * 16 + r + 8 * half;
            float val = o[f][r] / lrow[r];
            heads[((size_t)b * SEQ + s) * DM + hh * 64 + f * 16 + l16] = (_Float16)val;
        }
    }
}

// ---------------------------------------------------------------------------
// Output projection: out = heads(16384x512 f16) @ Wo + bo (Wo transposed).
// ---------------------------------------------------------------------------
__global__ __launch_bounds__(128)
void gemm_o(const _Float16* __restrict__ ah, const _Float16* __restrict__ wt,
            const float* __restrict__ bo, float* __restrict__ out) {
    __shared__ __align__(16) _Float16 x_lds[2][64][32];
    __shared__ __align__(16) _Float16 w_lds[2][64][32];

    const int row0 = blockIdx.x * 64;
    const int cb   = blockIdx.y;
    const int t = threadIdx.x, lane = t & 31, w = t >> 5;
    const int half = lane >> 4, l16 = lane & 15;

    const int rs = t >> 1, cs = (t & 1) * 16;

    auto stage = [&](int kb, int buf) {
        const _Float16* gx = ah + (size_t)(row0 + rs) * DM + kb * 32 + cs;
        _Float16* sx = &x_lds[buf][rs][cs];
        async_cp16(sx, gx); async_cp16(sx + 8, gx + 8);
        const _Float16* gw = wt + (size_t)(cb * 64 + rs) * DM + kb * 32 + cs;
        _Float16* sw = &w_lds[buf][rs][cs];
        async_cp16(sw, gw); async_cp16(sw + 8, gw + 8);
    };

    v8f acc[4] = {};

    stage(0, 0);
    async_wait();
    __syncthreads();
    stage(1, 1);

    for (int kb = 0; kb < 16; ++kb) {
        const int buf = kb & 1;
        v16h a;
        #pragma unroll
        for (int h = 0; h < 16; ++h) a[h] = x_lds[buf][w * 16 + l16][kmap(h, half)];
        #pragma unroll
        for (int f = 0; f < 4; ++f) {
            v16h b;
            #pragma unroll
            for (int h = 0; h < 16; ++h) b[h] = w_lds[buf][f * 16 + l16][kmap(h, half)];
            acc[f] = __builtin_amdgcn_wmma_f32_16x16x32_f16(
                false, a, false, b, (short)0, acc[f], false, false);
        }
        if (kb < 15) {
            async_wait();
            __syncthreads();
            if (kb < 14) stage(kb + 2, buf);
        }
    }

    #pragma unroll
    for (int f = 0; f < 4; ++f) {
        int c = cb * 64 + f * 16 + l16;
        float bsv = bo[c];
        #pragma unroll
        for (int r = 0; r < 8; ++r) {
            int row = row0 + w * 16 + r + 8 * half;
            out[(size_t)row * DM + c] = acc[f][r] + bsv;
        }
    }
}

extern "C" void kernel_launch(void* const* d_in, const int* in_sizes, int n_in,
                              void* d_out, int out_size, void* d_ws, size_t ws_size,
                              hipStream_t stream) {
    const float* x     = (const float*)d_in[0];
    const int*   pmask = (const int*)d_in[1];
    const float* wq = (const float*)d_in[2];
    const float* bq = (const float*)d_in[3];
    const float* wk = (const float*)d_in[4];
    const float* bk = (const float*)d_in[5];
    const float* wv = (const float*)d_in[6];
    const float* bv = (const float*)d_in[7];
    const float* wo = (const float*)d_in[8];
    const float* bo = (const float*)d_in[9];

    char* ws = (char*)d_ws;
    size_t off = 0;
    const size_t XH_B = (size_t)BATCH * SEQ * DM * 2;         // 16 MB
    const size_t W_B  = (size_t)DM * DM * 2;                  // 0.5 MB
    const size_t HM_B = (size_t)NB * SEQ * HD * 2;            // 16 MB

    _Float16* xh  = (_Float16*)(ws + off); off += XH_B;
    _Float16* wqt = (_Float16*)(ws + off); off += W_B;
    _Float16* wkt = (_Float16*)(ws + off); off += W_B;
    _Float16* wvt = (_Float16*)(ws + off); off += W_B;
    _Float16* wot = (_Float16*)(ws + off); off += W_B;
    _Float16* Qh  = (_Float16*)(ws + off); off += HM_B;
    _Float16* Kh  = (_Float16*)(ws + off); off += HM_B;
    _Float16* VhT = (_Float16*)(ws + off); off += HM_B;
    _Float16* heads = (_Float16*)(ws + off); off += XH_B;

    float* out = (float*)d_out;
    float* raw = out + (size_t)BATCH * SEQ * DM;   // raw_score after `out`

    cvt_f32_f16<<<2048, 256, 0, stream>>>(x, xh, BATCH * SEQ * DM);
    cvt_t_f32_f16<<<256, 256, 0, stream>>>(wq, wqt);
    cvt_t_f32_f16<<<256, 256, 0, stream>>>(wk, wkt);
    cvt_t_f32_f16<<<256, 256, 0, stream>>>(wv, wvt);
    cvt_t_f32_f16<<<256, 256, 0, stream>>>(wo, wot);

    gemm_qkv<<<dim3(256, 8, 3), 128, 0, stream>>>(xh, wqt, wkt, wvt,
                                                  bq, bk, bv, Qh, Kh, VhT);
    attn<<<dim3(SEQ / 64, NB), 128, 0, stream>>>(Qh, Kh, VhT, pmask, raw, heads);
    gemm_o<<<dim3(256, 8), 128, 0, stream>>>(heads, wot, bo, out);
}
